// SFGAT_47382079209645
// MI455X (gfx1250) — matrix operand order repaired and checked by reading, not compiled, not run on hardware
//
#include <hip/hip_runtime.h>
#include <stdint.h>

#define N_NODES 6000
#define NFEAT   512
#define NH      64
#define NPAD    6016          // padded node count (multiple of 32)
#define NWORDS  188           // ceil(6000/32)
#define NCHUNK  188
#define SEGS    8
#define CPS     24            // chunks per segment: ceil(188/8)
#define ALPHA   0.2f
#define NEGINF  (-9e15f)

typedef __attribute__((ext_vector_type(16))) __bf16 v16bf;
typedef __attribute__((ext_vector_type(8)))  float  v8f;

union ABPack { uint4 q[2]; uint32_t w[8]; v16bf v; };

__device__ __forceinline__ uint16_t f2bf(float f) {
  union { float f; uint32_t u; } x; x.f = f;
  return (uint16_t)((x.u + 0x7fffu + ((x.u >> 16) & 1u)) >> 16);
}
__device__ __forceinline__ float bf2f(uint16_t h) {
  union { uint32_t u; float f; } x; x.u = ((uint32_t)h) << 16;
  return x.f;
}
__device__ __forceinline__ v8f wmma_bf16(v16bf a, v16bf b, v8f c) {
  return __builtin_amdgcn_wmma_f32_16x16x32_bf16(false, a, false, b, (short)0, c,
                                                 false, false);
}

// ---------------- prep kernels ----------------

__global__ void k_f2bf(const float* __restrict__ in, uint16_t* __restrict__ out, int n) {
  int t = blockIdx.x * blockDim.x + threadIdx.x;
  if (t < n) out[t] = f2bf(in[t]);
}

// w: [K,64] f32 row-major  ->  wt: [64,K] bf16 (output-feature-major)
__global__ void k_wT(const float* __restrict__ w, uint16_t* __restrict__ wt, int K) {
  int t = blockIdx.x * blockDim.x + threadIdx.x;
  if (t >= 64 * K) return;
  int f = t / K, k = t % K;
  wt[t] = f2bf(w[k * 64 + f]);
}

// adjacency int32 [N,N] -> bit mask [NPAD, NWORDS]; pad rows/cols give 0 bits
__global__ void k_pack(const int* __restrict__ adj, uint32_t* __restrict__ bits) {
  long long t = (long long)blockIdx.x * blockDim.x + threadIdx.x;
  long long word = t >> 5;
  int lane = (int)(t & 31);
  if (word >= (long long)NPAD * NWORDS) return;
  int row = (int)(word / NWORDS);
  int w   = (int)(word % NWORDS);
  int j   = w * 32 + lane;
  int pred = 0;
  if (row < N_NODES && j < N_NODES) pred = adj[(size_t)row * N_NODES + j] > 0;
  unsigned long long m = __ballot(pred);
  if (lane == 0) bits[word] = (uint32_t)m;
}

__global__ void k_zpad(uint16_t* __restrict__ wht) {
  int t = blockIdx.x * blockDim.x + threadIdx.x;
  if (t >= 64 * 16) return;
  int n = t >> 4, c = t & 15;
  wht[(size_t)n * NPAD + N_NODES + c] = 0;
}

// ---------------- GEMM: Wh^T = (A[6000,K] @ WT^T) transposed, bf16 ----------------
// A bf16 row-major (lda=K), WT bf16 [64,K] feature-major. Output WhT bf16 [64,NPAD].
__global__ __launch_bounds__(32) void k_gemm(const uint16_t* __restrict__ A, int K,
                                             const uint16_t* __restrict__ WT,
                                             uint16_t* __restrict__ WhT) {
  const int lane = threadIdx.x, li = lane & 15;
  const bool lo = lane < 16;
  const int row = blockIdx.x * 16 + li;
  const uint16_t* arow = A + (size_t)row * K;
  v8f acc0 = {}, acc1 = {}, acc2 = {}, acc3 = {};
  for (int k0 = 0; k0 < K; k0 += 32) {
    ABPack Av;
    Av.q[0] = *(const uint4*)(arow + k0 + (lo ? 0 : 16));       // K 0..7 / 16..23
    Av.q[1] = *(const uint4*)(arow + k0 + 8 + (lo ? 0 : 16));   // K 8..15 / 24..31
    const int kbb = k0 + (lo ? 0 : 16);
    ABPack B0, B1, B2, B3;
    { const uint16_t* p = WT + (size_t)(li)      * K + kbb;
      B0.q[0] = ((const uint4*)p)[0]; B0.q[1] = ((const uint4*)p)[1]; }
    { const uint16_t* p = WT + (size_t)(li + 16) * K + kbb;
      B1.q[0] = ((const uint4*)p)[0]; B1.q[1] = ((const uint4*)p)[1]; }
    { const uint16_t* p = WT + (size_t)(li + 32) * K + kbb;
      B2.q[0] = ((const uint4*)p)[0]; B2.q[1] = ((const uint4*)p)[1]; }
    { const uint16_t* p = WT + (size_t)(li + 48) * K + kbb;
      B3.q[0] = ((const uint4*)p)[0]; B3.q[1] = ((const uint4*)p)[1]; }
    acc0 = wmma_bf16(Av.v, B0.v, acc0);
    acc1 = wmma_bf16(Av.v, B1.v, acc1);
    acc2 = wmma_bf16(Av.v, B2.v, acc2);
    acc3 = wmma_bf16(Av.v, B3.v, acc3);
  }
  // store transposed: WhT[n][blockRow0 + M], 8 consecutive rows per lane -> one b128
  const int cb = blockIdx.x * 16 + (lo ? 0 : 8);
#define STORE_T(ACC, T)                                                        \
  {                                                                            \
    uint32_t wd[4];                                                            \
    _Pragma("unroll") for (int r = 0; r < 4; ++r)                              \
      wd[r] = (uint32_t)f2bf(ACC[2 * r]) | ((uint32_t)f2bf(ACC[2 * r + 1]) << 16); \
    *(uint4*)(WhT + (size_t)(li + 16 * T) * NPAD + cb) =                       \
        make_uint4(wd[0], wd[1], wd[2], wd[3]);                                \
  }
  STORE_T(acc0, 0) STORE_T(acc1, 1) STORE_T(acc2, 2) STORE_T(acc3, 3)
#undef STORE_T
}

// f[j] = Wh[j]·a[0:64], g[j] = Wh[j]·a[64:128]  (reads WhT columns, padded->0)
__global__ void k_fg(const uint16_t* __restrict__ WhT, const float* __restrict__ a,
                     float* __restrict__ f, float* __restrict__ g) {
  int j = blockIdx.x * blockDim.x + threadIdx.x;
  if (j >= NPAD) return;
  if (j >= N_NODES) { f[j] = 0.0f; g[j] = 0.0f; return; }
  float sf = 0.0f, sg = 0.0f;
#pragma unroll
  for (int n = 0; n < 64; ++n) {
    float w = bf2f(WhT[(size_t)n * NPAD + j]);
    sf += w * a[n];
    sg += w * a[64 + n];
  }
  f[j] = sf; g[j] = sg;
}

// ---------------- flash masked-softmax aggregation ----------------
// grid (188, SEGS), 1 wave per block, 32 rows per block (two 16-row P tiles
// sharing the same streamed B tiles -> halves L2 traffic per row).
__global__ __launch_bounds__(32) void k_flash(const float* __restrict__ fv,
                                              const float* __restrict__ gv,
                                              const uint16_t* __restrict__ WhT,
                                              const uint32_t* __restrict__ bits,
                                              float* __restrict__ Pm,
                                              float* __restrict__ Pl,
                                              float* __restrict__ Pacc) {
  const int bx = blockIdx.x, s = blockIdx.y;
  const int lane = threadIdx.x, li = lane & 15;
  const bool lo = lane < 16;
  const int r0 = bx * 32 + li;        // group-0 row of this lane
  const int r1 = r0 + 16;             // group-1 row
  const float fi0 = fv[r0];
  const float fi1 = fv[r1];
  const int kb = lo ? 0 : 16;

  v8f a00 = {}, a01 = {}, a02 = {}, a03 = {};   // rows 0..15 of block
  v8f a10 = {}, a11 = {}, a12 = {}, a13 = {};   // rows 16..31 of block
  float m0 = NEGINF, l0 = 0.0f, m1 = NEGINF, l1 = 0.0f;

  const int cbeg = s * CPS;
  const int cend = min(cbeg + CPS, NCHUNK);
  const uint32_t* brow0 = bits + (size_t)r0 * NWORDS;
  const uint32_t* brow1 = bits + (size_t)r1 * NWORDS;

  for (int ch = cbeg; ch < cend; ++ch) {
    const int j0 = ch * 32;
    float4 gA = *(const float4*)(gv + j0 + kb);
    float4 gB = *(const float4*)(gv + j0 + kb + 4);
    float4 gC = *(const float4*)(gv + j0 + 8 + kb);
    float4 gD = *(const float4*)(gv + j0 + 8 + kb + 4);
    float gval[16] = {gA.x, gA.y, gA.z, gA.w, gB.x, gB.y, gB.z, gB.w,
                      gC.x, gC.y, gC.z, gC.w, gD.x, gD.y, gD.z, gD.w};
    const uint32_t mk0 = brow0[ch];
    const uint32_t mk1 = brow1[ch];

    float e0[16], e1[16];
    float cm0 = NEGINF, cm1 = NEGINF;
#pragma unroll
    for (int c = 0; c < 16; ++c) {
      const int K = c + kb;   // lane-half A-operand K positions
      float t0 = fi0 + gval[c];
      t0 = (t0 > 0.0f) ? t0 : ALPHA * t0;
      t0 = ((mk0 >> K) & 1u) ? t0 : NEGINF;
      e0[c] = t0; cm0 = fmaxf(cm0, t0);
      float t1 = fi1 + gval[c];
      t1 = (t1 > 0.0f) ? t1 : ALPHA * t1;
      t1 = ((mk1 >> K) & 1u) ? t1 : NEGINF;
      e1[c] = t1; cm1 = fmaxf(cm1, t1);
    }
    cm0 = fmaxf(cm0, __shfl_xor(cm0, 16, 32));  // combine the two K-halves of a row
    cm1 = fmaxf(cm1, __shfl_xor(cm1, 16, 32));
    const float mn0 = fmaxf(m0, cm0);
    const float mn1 = fmaxf(m1, cm1);

    float cs0 = 0.0f, cs1 = 0.0f;
    ABPack P0, P1;
#pragma unroll
    for (int c = 0; c < 16; c += 2) {
      float p0a = __expf(e0[c] - mn0), p0b = __expf(e0[c + 1] - mn0);
      cs0 += p0a + p0b;
      P0.w[c >> 1] = (uint32_t)f2bf(p0a) | ((uint32_t)f2bf(p0b) << 16);
      float p1a = __expf(e1[c] - mn1), p1b = __expf(e1[c + 1] - mn1);
      cs1 += p1a + p1b;
      P1.w[c >> 1] = (uint32_t)f2bf(p1a) | ((uint32_t)f2bf(p1b) << 16);
    }
    cs0 += __shfl_xor(cs0, 16, 32);
    cs1 += __shfl_xor(cs1, 16, 32);
    const float scale0 = __expf(m0 - mn0);
    const float scale1 = __expf(m1 - mn1);
    l0 = l0 * scale0 + cs0;
    l1 = l1 * scale1 + cs1;
    const bool chg = (mn0 > m0) | (mn1 > m1);
    m0 = mn0; m1 = mn1;

    // rescale accumulators only if some row max advanced (wave-uniform branch,
    // EXEC stays all-ones for the WMMAs below)
    if (__ballot(chg) != 0ull) {
#pragma unroll
      for (int r = 0; r < 8; ++r) {
        const float sr0 = __shfl(scale0, r + (lo ? 0 : 8), 32);
        const float sr1 = __shfl(scale1, r + (lo ? 0 : 8), 32);
        a00[r] *= sr0; a01[r] *= sr0; a02[r] *= sr0; a03[r] *= sr0;
        a10[r] *= sr1; a11[r] *= sr1; a12[r] *= sr1; a13[r] *= sr1;
      }
    }

    // B operands: WhT[n][j0 + Krange], contiguous 32B per lane, shared by both P tiles
    const uint16_t* bb = WhT + j0 + (lo ? 0 : 16);
    ABPack B0, B1, B2, B3;
    { const uint16_t* p = bb + (size_t)(li)      * NPAD;
      B0.q[0] = ((const uint4*)p)[0]; B0.q[1] = ((const uint4*)p)[1]; }
    { const uint16_t* p = bb + (size_t)(li + 16) * NPAD;
      B1.q[0] = ((const uint4*)p)[0]; B1.q[1] = ((const uint4*)p)[1]; }
    { const uint16_t* p = bb + (size_t)(li + 32) * NPAD;
      B2.q[0] = ((const uint4*)p)[0]; B2.q[1] = ((const uint4*)p)[1]; }
    { const uint16_t* p = bb + (size_t)(li + 48) * NPAD;
      B3.q[0] = ((const uint4*)p)[0]; B3.q[1] = ((const uint4*)p)[1]; }

    a00 = wmma_bf16(P0.v, B0.v, a00);
    a01 = wmma_bf16(P0.v, B1.v, a01);
    a02 = wmma_bf16(P0.v, B2.v, a02);
    a03 = wmma_bf16(P0.v, B3.v, a03);
    a10 = wmma_bf16(P1.v, B0.v, a10);
    a11 = wmma_bf16(P1.v, B1.v, a11);
    a12 = wmma_bf16(P1.v, B2.v, a12);
    a13 = wmma_bf16(P1.v, B3.v, a13);
  }

  const int ob = bx * 32 + (lo ? 0 : 8);
#pragma unroll
  for (int r = 0; r < 8; ++r) {
    float* d0 = Pacc + ((size_t)s * NPAD + ob + r) * NH;
    d0[li]      = a00[r];
    d0[li + 16] = a01[r];
    d0[li + 32] = a02[r];
    d0[li + 48] = a03[r];
    float* d1 = Pacc + ((size_t)s * NPAD + ob + 16 + r) * NH;
    d1[li]      = a10[r];
    d1[li + 16] = a11[r];
    d1[li + 32] = a12[r];
    d1[li + 48] = a13[r];
  }
  if (lo) {
    Pm[(size_t)s * NPAD + r0] = m0;
    Pl[(size_t)s * NPAD + r0] = l0;
    Pm[(size_t)s * NPAD + r1] = m1;
    Pl[(size_t)s * NPAD + r1] = l1;
  }
}

// ---------------- combines ----------------

// layer-1: merge segments, normalize, elu -> h1 bf16 row-major
__global__ __launch_bounds__(256) void k_comb1(const float* __restrict__ Pm,
                                               const float* __restrict__ Pl,
                                               const float* __restrict__ Pacc,
                                               uint16_t* __restrict__ h1) {
  const int bx = blockIdx.x, t = threadIdx.x;
  const int r = t >> 4, c0 = (t & 15) * 4;
  const int row = bx * 16 + r;
  float ms[SEGS], ws[SEGS];
  float M = NEGINF;
#pragma unroll
  for (int s = 0; s < SEGS; ++s) { ms[s] = Pm[(size_t)s * NPAD + row]; M = fmaxf(M, ms[s]); }
  float L = 0.0f;
#pragma unroll
  for (int s = 0; s < SEGS; ++s) { ws[s] = __expf(ms[s] - M); L += Pl[(size_t)s * NPAD + row] * ws[s]; }
  const float inv = 1.0f / L;
#pragma unroll
  for (int c = c0; c < c0 + 4; ++c) {
    float v = 0.0f;
#pragma unroll
    for (int s = 0; s < SEGS; ++s)
      v += Pacc[((size_t)s * NPAD + row) * NH + c] * ws[s];
    float h = v * inv;
    float el = (h > 0.0f) ? h : (__expf(h) - 1.0f);
    h1[(size_t)row * NH + c] = f2bf(el);
  }
}

// layer-2: merge segments, normalize, elu, row log_softmax -> f32 output slot
__global__ __launch_bounds__(256) void k_comb2(const float* __restrict__ Pm,
                                               const float* __restrict__ Pl,
                                               const float* __restrict__ Pacc,
                                               float* __restrict__ outp) {
  const int bx = blockIdx.x, t = threadIdx.x;
  const int r = t >> 4, li = t & 15, c0 = li * 4;
  const int row = bx * 16 + r;
  float ms[SEGS], ws[SEGS];
  float M = NEGINF;
#pragma unroll
  for (int s = 0; s < SEGS; ++s) { ms[s] = Pm[(size_t)s * NPAD + row]; M = fmaxf(M, ms[s]); }
  float L = 0.0f;
#pragma unroll
  for (int s = 0; s < SEGS; ++s) { ws[s] = __expf(ms[s] - M); L += Pl[(size_t)s * NPAD + row] * ws[s]; }
  const float inv = 1.0f / L;
  float y[4];
#pragma unroll
  for (int k = 0; k < 4; ++k) {
    float v = 0.0f;
#pragma unroll
    for (int s = 0; s < SEGS; ++s)
      v += Pacc[((size_t)s * NPAD + row) * NH + c0 + k] * ws[s];
    float h = v * inv;
    y[k] = (h > 0.0f) ? h : (__expf(h) - 1.0f);   // elu
  }
  // log_softmax over the row's 64 values (16 lanes x 4, xor-reduce within 16-group)
  float my = fmaxf(fmaxf(y[0], y[1]), fmaxf(y[2], y[3]));
#pragma unroll
  for (int d = 1; d < 16; d <<= 1) my = fmaxf(my, __shfl_xor(my, d, 32));
  float se = __expf(y[0] - my) + __expf(y[1] - my) + __expf(y[2] - my) + __expf(y[3] - my);
#pragma unroll
  for (int d = 1; d < 16; d <<= 1) se += __shfl_xor(se, d, 32);
  const float lse = __logf(se);
#pragma unroll
  for (int k = 0; k < 4; ++k)
    outp[(size_t)row * NH + c0 + k] = y[k] - my - lse;
}

// ---------------- fusion head ----------------
__global__ __launch_bounds__(128) void k_fuse(const float* __restrict__ e1,
                                              const float* __restrict__ e2,
                                              const float* __restrict__ c1,
                                              const float* __restrict__ c2,
                                              const float* __restrict__ aw1,
                                              const float* __restrict__ ab1,
                                              const float* __restrict__ aw2,
                                              const float* __restrict__ mw,
                                              const float* __restrict__ mb,
                                              float* __restrict__ outp,
                                              float* __restrict__ betap,
                                              float* __restrict__ embp) {
  const int i = blockIdx.x * blockDim.x + threadIdx.x;
  if (i >= N_NODES) return;
  const float* z0 = e1 + (size_t)i * NH;
  const float* z1 = e2 + (size_t)i * NH;
  const float* x1 = c1 + (size_t)i * NH;
  const float* x2 = c2 + (size_t)i * NH;

  float w[3];
#pragma unroll
  for (int s = 0; s < 3; ++s) {
    float a16[16];
#pragma unroll
    for (int h = 0; h < 16; ++h) a16[h] = ab1[h];
    for (int n = 0; n < NH; ++n) {
      float zv = (s == 0) ? z0[n] : (s == 1) ? z1[n] : 0.5f * (x1[n] + x2[n]);
#pragma unroll
      for (int h = 0; h < 16; ++h) a16[h] += zv * aw1[n * 16 + h];
    }
    float acc = 0.0f;
#pragma unroll
    for (int h = 0; h < 16; ++h) acc += tanhf(a16[h]) * aw2[h];
    w[s] = acc;
  }
  float mx = fmaxf(w[0], fmaxf(w[1], w[2]));
  float b[3], sm = 0.0f;
#pragma unroll
  for (int s = 0; s < 3; ++s) { b[s] = __expf(w[s] - mx); sm += b[s]; }
#pragma unroll
  for (int s = 0; s < 3; ++s) { b[s] /= sm; betap[(size_t)i * 3 + s] = b[s]; }

  float o[8];
#pragma unroll
  for (int c = 0; c < 8; ++c) o[c] = mb[c];
  for (int n = 0; n < NH; ++n) {
    float e = b[0] * z0[n] + b[1] * z1[n] + b[2] * 0.5f * (x1[n] + x2[n]);
    embp[(size_t)i * NH + n] = e;
#pragma unroll
    for (int c = 0; c < 8; ++c) o[c] += e * mw[n * 8 + c];
  }
  float mo = o[0];
#pragma unroll
  for (int c = 1; c < 8; ++c) mo = fmaxf(mo, o[c]);
  float so = 0.0f;
#pragma unroll
  for (int c = 0; c < 8; ++c) so += __expf(o[c] - mo);
  const float ls = __logf(so);
#pragma unroll
  for (int c = 0; c < 8; ++c) outp[(size_t)i * 8 + c] = o[c] - mo - ls;
}

// ---------------- host launcher ----------------
extern "C" void kernel_launch(void* const* d_in, const int* in_sizes, int n_in,
                              void* d_out, int out_size, void* d_ws, size_t ws_size,
                              hipStream_t stream) {
  (void)in_sizes; (void)n_in; (void)out_size; (void)ws_size;
  const float* x    = (const float*)d_in[0];
  const int*   sadj = (const int*)d_in[1];
  const int*   fadj = (const int*)d_in[2];
  const float* wset[3][4] = {
    {(const float*)d_in[3],  (const float*)d_in[4],  (const float*)d_in[5],  (const float*)d_in[6]},   // s1
    {(const float*)d_in[7],  (const float*)d_in[8],  (const float*)d_in[9],  (const float*)d_in[10]},  // s2
    {(const float*)d_in[11], (const float*)d_in[12], (const float*)d_in[13], (const float*)d_in[14]},  // c
  };
  const float* aw1 = (const float*)d_in[15];
  const float* ab1 = (const float*)d_in[16];
  const float* aw2 = (const float*)d_in[17];
  const float* mw  = (const float*)d_in[18];
  const float* mb  = (const float*)d_in[19];
  float* out = (float*)d_out;

  char* ws = (char*)d_ws;
  uint16_t* Xbf   = (uint16_t*)(ws + 0);                          // 6,144,000
  uint16_t* W1T   = (uint16_t*)(ws + 6144000);                    // 3 * 65,536
  uint16_t* W2T   = (uint16_t*)(ws + 6340608);                    // 3 * 8,192
  uint32_t* BITS0 = (uint32_t*)(ws + 6365184);                    // 4,524,032 (NPAD rows)
  uint32_t* BITS1 = (uint32_t*)(ws + 10889216);                   // 4,524,032
  uint16_t* WhT   = (uint16_t*)(ws + 15413248);                   // 770,048
  float*    Fv    = (float*)(ws + 16183296);                      // 24,064
  float*    Gv    = (float*)(ws + 16207360);                      // 24,064
  uint16_t* H1    = (uint16_t*)(ws + 16231424);                   // 768,000
  float*    Pm    = (float*)(ws + 16999424);                      // 192,512 (NPAD rows)
  float*    Pl    = (float*)(ws + 17191936);                      // 192,512
  float*    Pacc  = (float*)(ws + 17384448);                      // 12,320,768 (NPAD rows)

  // output slots (float element offsets)
  float* OUT  = out + 0;
  float* BETA = out + 48000;
  float* EMB1 = out + 66000;
  float* COM1 = out + 450000;
  float* COM2 = out + 834000;
  float* EMB2 = out + 1218000;
  float* EMB  = out + 1602000;

  // ---- prep ----
  k_f2bf<<<(N_NODES * NFEAT + 255) / 256, 256, 0, stream>>>(x, Xbf, N_NODES * NFEAT);
  for (int s = 0; s < 3; ++s) {
    k_wT<<<(64 * NFEAT + 255) / 256, 256, 0, stream>>>(wset[s][0], W1T + (size_t)s * 64 * NFEAT, NFEAT);
    k_wT<<<(64 * NH + 255) / 256, 256, 0, stream>>>(wset[s][2], W2T + (size_t)s * 64 * NH, NH);
  }
  {
    long long tot = (long long)NPAD * NWORDS * 32;
    int blocks = (int)((tot + 255) / 256);
    k_pack<<<blocks, 256, 0, stream>>>(sadj, BITS0);
    k_pack<<<blocks, 256, 0, stream>>>(fadj, BITS1);
  }
  k_zpad<<<(64 * 16 + 255) / 256, 256, 0, stream>>>(WhT);

  // ---- 4 GAT runs: (weight set, adjacency bits, output slot) ----
  const int     runset[4]  = {0, 2, 2, 1};                // s1, c, c, s2
  uint32_t*     runbits[4] = {BITS0, BITS0, BITS1, BITS1};
  float*        runout[4]  = {EMB1, COM1, COM2, EMB2};

  const int RB16 = N_NODES / 16;                          // 375 GEMM/combine blocks
  const int RB32 = NPAD / 32;                             // 188 flash row blocks
  for (int r = 0; r < 4; ++r) {
    const int si = runset[r];
    uint16_t* w1t = W1T + (size_t)si * 64 * NFEAT;
    uint16_t* w2t = W2T + (size_t)si * 64 * NH;
    // layer 1
    k_gemm<<<RB16, 32, 0, stream>>>(Xbf, NFEAT, w1t, WhT);
    k_fg<<<(NPAD + 255) / 256, 256, 0, stream>>>(WhT, wset[si][1], Fv, Gv);
    k_flash<<<dim3(RB32, SEGS), 32, 0, stream>>>(Fv, Gv, WhT, runbits[r], Pm, Pl, Pacc);
    k_comb1<<<RB16, 256, 0, stream>>>(Pm, Pl, Pacc, H1);
    // layer 2
    k_gemm<<<RB16, 32, 0, stream>>>(H1, NH, w2t, WhT);
    k_fg<<<(NPAD + 255) / 256, 256, 0, stream>>>(WhT, wset[si][3], Fv, Gv);
    k_flash<<<dim3(RB32, SEGS), 32, 0, stream>>>(Fv, Gv, WhT, runbits[r], Pm, Pl, Pacc);
    k_comb2<<<RB16, 256, 0, stream>>>(Pm, Pl, Pacc, runout[r]);
  }

  // ---- fusion head ----
  k_fuse<<<(N_NODES + 127) / 128, 128, 0, stream>>>(EMB1, EMB2, COM1, COM2,
                                                    aw1, ab1, aw2, mw, mb,
                                                    OUT, BETA, EMB);
}
